// KMeansProxy_32418413150285
// MI455X (gfx1250) — compile-verified, matrix operand order checked by reading, not run on hardware
//
#include <hip/hip_runtime.h>

#define NPTS 32768
#define KP   4000
#define DD   128
#define CC   1000
#define NTILES (KP / 16)          // 250 proxy tiles of 16
#define ROWS_US (256 + 8)         // LDS row stride in ushorts: 512B data + 16B pad = 528B

typedef __attribute__((ext_vector_type(16))) __bf16          v16bf;
typedef __attribute__((ext_vector_type(8)))  __bf16          v8bf;
typedef __attribute__((ext_vector_type(16))) unsigned short  v16us;
typedef __attribute__((ext_vector_type(8)))  float           v8f;
typedef __attribute__((ext_vector_type(4)))  unsigned int    v4u;
typedef __attribute__((ext_vector_type(8)))  int             v8i;
typedef __attribute__((ext_vector_type(4)))  int             v4i;

// Split an f32 into bf16 hi (RNE) + bf16 lo (residual, RNE).
__device__ __forceinline__ void f32_to_bf16_pair(float v, unsigned short& h, unsigned short& l) {
    unsigned int u  = __float_as_uint(v);
    unsigned int r  = u + 0x7FFFu + ((u >> 16) & 1u);
    h = (unsigned short)(r >> 16);
    float fh = __uint_as_float(((unsigned int)h) << 16);
    float lo = v - fh;
    unsigned int u2 = __float_as_uint(lo);
    unsigned int r2 = u2 + 0x7FFFu + ((u2 >> 16) & 1u);
    l = (unsigned short)(r2 >> 16);
}

__device__ __forceinline__ v16bf combine8(v8bf a, v8bf b) {
    v16bf r;
#pragma unroll
    for (int i = 0; i < 8; ++i) { r[i] = a[i]; r[i + 8] = b[i]; }
    return r;
}

// ---------------------------------------------------------------------------
// TDM: DMA one proxy tile (16 rows x 256 bf16 = 8KB) from global to LDS.
// 2D descriptor per CDNA5 ISA sec.8; pad 4 dwords (16B)
// after every 128 dwords (one 512B row) -> LDS row stride 528B (bank-spread).
// ---------------------------------------------------------------------------
__device__ __forceinline__ void tdm_load_tile(const unsigned short* gsrc, unsigned int lds_off) {
    unsigned long long ga = (unsigned long long)(uintptr_t)gsrc;
    v4u g0;
    g0[0] = 1u;                                            // count=1 (valid user D#)
    g0[1] = lds_off;                                       // lds_addr (bytes)
    g0[2] = (unsigned int)(ga & 0xFFFFFFFFu);              // global_addr[31:0]
    g0[3] = (unsigned int)((ga >> 32) & 0x01FFFFFFu)       // global_addr[56:32]
          | (2u << 30);                                    // type = 2 ("image")
    v8i g1;
    g1[0] = (int)((1u << 16)                               // data_size = 1 -> 2 bytes
                | (1u << 20)                               // pad_enable
                | (6u << 22)                               // pad_interval: 128 dwords
                | (3u << 25));                             // pad_amount: 4 dwords (16B)
    g1[1] = (int)(256u << 16);                             // tensor_dim0 = 256 (lo16)
    g1[2] = (int)(16u << 16);                              // tensor_dim0 hi=0 | tensor_dim1=16
    g1[3] = (int)(256u << 16);                             // tensor_dim1 hi=0 | tile_dim0=256
    g1[4] = 16;                                            // tile_dim1=16, tile_dim2=0
    g1[5] = 256;                                           // tensor_dim0_stride lo32 = 256
    g1[6] = 0;                                             // stride hi | dim1_stride lo
    g1[7] = 0;
    v4i z = {0, 0, 0, 0};
#if defined(__clang_major__) && (__clang_major__ >= 23)
    v8i z8 = {0, 0, 0, 0, 0, 0, 0, 0};
    __builtin_amdgcn_tensor_load_to_lds(g0, g1, z, z, z8, 0);
#else
    __builtin_amdgcn_tensor_load_to_lds(g0, g1, z, z, 0);
#endif
}

// ---------------------------------------------------------------------------
// Kernel 1: p2[k] = sum(p^2); combined bf16 row  pcomb[k] = [hi(128) | lo(128)].
// ---------------------------------------------------------------------------
__global__ __launch_bounds__(128) void prep_kernel(const float* __restrict__ proxies,
                                                   unsigned short* __restrict__ pcomb,
                                                   float* __restrict__ p2) {
    int k = blockIdx.x;           // one proxy row per block
    int t = threadIdx.x;          // 128 threads == D
    float v = proxies[(size_t)k * DD + t];
    unsigned short h, l;
    f32_to_bf16_pair(v, h, l);
    pcomb[(size_t)k * 256 + t]       = h;
    pcomb[(size_t)k * 256 + 128 + t] = l;

    float sq = v * v;
#pragma unroll
    for (int off = 16; off > 0; off >>= 1) sq += __shfl_down(sq, off, 32);
    __shared__ float red[4];
    int wave = t >> 5, lane = t & 31;
    if (lane == 0) red[wave] = sq;
    __syncthreads();
    if (t == 0) p2[k] = red[0] + red[1] + red[2] + red[3];
}

// ---------------------------------------------------------------------------
// Kernel 2: argmax_k ( x.p_k - 0.5*p2[k] )  via split-bf16 WMMA.
// Block = 128 threads (4 waves). Proxy tiles stream global->LDS with the
// Tensor Data Mover, double buffered (TENSORcnt + workgroup barrier); all
// 4 waves share each LDS tile. x fragments (B operand) live in registers.
// All 8 A fragments are loaded from LDS up front so the 12 WMMAs issue
// back-to-back with a single dscnt wait; argmax tail is branchless.
// ---------------------------------------------------------------------------
__global__ __launch_bounds__(128) void assign_kernel(const float* __restrict__ x,
                                                     const unsigned short* __restrict__ pcomb,
                                                     const float* __restrict__ p2,
                                                     int* __restrict__ idx) {
    __shared__ __align__(16) unsigned short tileBuf[2][16 * ROWS_US];  // 2 x 8448B

    const int lane    = threadIdx.x & 31;
    const int wave    = threadIdx.x >> 5;
    const int rowBase = blockIdx.x * 64 + wave * 16;
    const int r       = lane & 15;   // x-row within tile == C column n; also A proxy row
    const int half    = lane >> 4;   // lane half
    const int mo      = half * 8;    // C row (proxy m) offset for this half

    // Prologue: DMA tile 0 into buffer 0 (wave 0 only; TDM ignores EXEC).
    if (wave == 0) {
        tdm_load_tile(pcomb, (unsigned int)(uintptr_t)&tileBuf[0][0]);
    }

    // Build x fragments (B matrix, 32x16 per K-step): lane holds x row `r`.
    const float* xrow = x + (size_t)(rowBase + r) * DD;
    v16bf xh[4], xl[4];
#pragma unroll
    for (int ks = 0; ks < 4; ++ks) {
        const int c0 = ks * 32 + half * 8;
        const int c1 = c0 + 16;
        v16us th, tl;
#pragma unroll
        for (int i = 0; i < 8; ++i) {
            unsigned short h, l;
            f32_to_bf16_pair(xrow[c0 + i], h, l);
            th[i] = h;  tl[i] = l;
            f32_to_bf16_pair(xrow[c1 + i], h, l);
            th[i + 8] = h;  tl[i + 8] = l;
        }
        xh[ks] = __builtin_bit_cast(v16bf, th);
        xl[ks] = __builtin_bit_cast(v16bf, tl);
    }

    float best  = -3.402823466e+38f;
    int   bestk = 0;

    for (int t = 0; t < NTILES; ++t) {
        const int k0  = t * 16;
        const int cur = t & 1;

        // Current tile's DMA must be complete before anyone reads LDS.
        if (wave == 0) __builtin_amdgcn_s_wait_tensorcnt(0);
        __syncthreads();

        // Kick off DMA for the next tile into the other buffer; it overlaps
        // the 12 WMMAs below.
        if (wave == 0 && t + 1 < NTILES) {
            tdm_load_tile(pcomb + (size_t)(k0 + 16) * 256,
                          (unsigned int)(uintptr_t)&tileBuf[cur ^ 1][0]);
        }

        // Seed accumulator with -0.5 * p2 for this tile's proxy rows.
        const float4* p2v = (const float4*)(p2 + k0 + mo);
        float4 pa = p2v[0], pb = p2v[1];
        v8f C;
        C[0] = -0.5f * pa.x; C[1] = -0.5f * pa.y; C[2] = -0.5f * pa.z; C[3] = -0.5f * pa.w;
        C[4] = -0.5f * pb.x; C[5] = -0.5f * pb.y; C[6] = -0.5f * pb.z; C[7] = -0.5f * pb.w;

        // Load ALL A fragments for this tile first (16 ds_load_b128 clause),
        // so the 12 WMMAs below run without per-step dscnt waits.
        const __bf16* tb = (const __bf16*)&tileBuf[cur][0];
        const int rowOff = r * ROWS_US;   // row stride 528B: hi at +0, lo at +128
        v16bf ah[4], al[4];
#pragma unroll
        for (int ks = 0; ks < 4; ++ks) {
            const int c0 = ks * 32 + half * 8;
            ah[ks] = combine8(*(const v8bf*)(tb + rowOff + c0),
                              *(const v8bf*)(tb + rowOff + c0 + 16));
            al[ks] = combine8(*(const v8bf*)(tb + rowOff + 128 + c0),
                              *(const v8bf*)(tb + rowOff + 128 + c0 + 16));
        }
#pragma unroll
        for (int ks = 0; ks < 4; ++ks) {
            // dot ~= p_hi.x_hi + p_hi.x_lo + p_lo.x_hi  (f32 accumulate)
            C = __builtin_amdgcn_wmma_f32_16x16x32_bf16(false, ah[ks], false, xh[ks], (short)0, C, false, false);
            C = __builtin_amdgcn_wmma_f32_16x16x32_bf16(false, ah[ks], false, xl[ks], (short)0, C, false, false);
            C = __builtin_amdgcn_wmma_f32_16x16x32_bf16(false, al[ks], false, xh[ks], (short)0, C, false, false);
        }

        // Branchless per-lane argmax over the 8 proxy rows held in C.
        // Ascending j with strict '>' keeps the lowest index on ties.
        float v  = C[0];
        int   vi = k0 + mo;
#pragma unroll
        for (int j = 1; j < 8; ++j) {
            float c = C[j]; int ci = k0 + j + mo;
            bool take = (c > v);
            v  = take ? c  : v;
            vi = take ? ci : vi;
        }
        // Merge across lane halves. Half 0 holds the lower proxy indices, so
        // on a tie half 0 keeps its own ('>'), half 1 takes the partner ('>=').
        float ov = __shfl_xor(v, 16, 32);
        int   oi = __shfl_xor(vi, 16, 32);
        {
            bool take = half ? (ov >= v) : (ov > v);
            v  = take ? ov : v;
            vi = take ? oi : vi;
        }
        // Fold into running best: earlier tiles have lower indices, so strict
        // '>' keeps the lowest index on ties.
        {
            bool take = (v > best);
            best  = take ? v  : best;
            bestk = take ? vi : bestk;
        }
    }

    if (half == 0) idx[rowBase + r] = bestk;
}

// ---------------------------------------------------------------------------
// Kernel 3: out = concat( x, proxies[idx], labels[idx] )  -- pure bandwidth.
// ---------------------------------------------------------------------------
__global__ __launch_bounds__(256) void gather_kernel(const float* __restrict__ x,
                                                     const float* __restrict__ proxies,
                                                     const float* __restrict__ labels,
                                                     const int* __restrict__ idx,
                                                     float* __restrict__ out) {
    const int n = blockIdx.x;
    const int t = threadIdx.x;
    const int k = idx[n];

    const float4* x4 = (const float4*)(x       + (size_t)n * DD);
    const float4* p4 = (const float4*)(proxies + (size_t)k * DD);
    const float4* l4 = (const float4*)(labels  + (size_t)k * CC);

    float4* ox = (float4*)(out + (size_t)n * DD);
    float4* op = (float4*)(out + (size_t)NPTS * DD + (size_t)n * DD);
    float4* ol = (float4*)(out + (size_t)NPTS * (2 * DD) + (size_t)n * CC);

    if (t < 32)       ox[t]      = x4[t];        // 128 f32 = 32 float4
    else if (t < 64)  op[t - 32] = p4[t - 32];   // 128 f32 = 32 float4
    for (int i = t; i < CC / 4; i += 256) ol[i] = l4[i];  // 1000 f32 = 250 float4
}

// ---------------------------------------------------------------------------
extern "C" void kernel_launch(void* const* d_in, const int* in_sizes, int n_in,
                              void* d_out, int out_size, void* d_ws, size_t ws_size,
                              hipStream_t stream) {
    const float* x       = (const float*)d_in[0];   // [N, D]
    const float* proxies = (const float*)d_in[1];   // [K, D]
    const float* labels  = (const float*)d_in[2];   // [K, C]
    float* out = (float*)d_out;

    // Workspace layout (~2.2 MB, all 16B-aligned):
    char* ws = (char*)d_ws;
    unsigned short* pcomb = (unsigned short*)ws;                  // K * 256 bf16 (hi|lo rows)
    float*          p2    = (float*)(pcomb + (size_t)KP * 256);   // K f32
    int*            idx   = (int*)(p2 + KP);                      // N i32

    prep_kernel  <<<KP,        128, 0, stream>>>(proxies, pcomb, p2);
    assign_kernel<<<NPTS / 64, 128, 0, stream>>>(x, pcomb, p2, idx);
    gather_kernel<<<NPTS,      256, 0, stream>>>(x, proxies, labels, idx, out);
    (void)in_sizes; (void)n_in; (void)out_size; (void)ws_size;
}
